// RNNLangModel_43052752175698
// MI455X (gfx1250) — compile-verified, matrix operand order checked by reading, not run on hardware
//
#include <hip/hip_runtime.h>

// ---------------------------------------------------------------------------
// RNN LM for MI455X (gfx1250, wave32, WMMA).
//   K1: P[t,b,:] = embed[input]·W1eT + b1      (f32 WMMA 16x16x4)
//   K2: persistent scan h' = sigmoid(h·W1hT+P) (f32 WMMA 16x16x4, LDS-staged
//       W1h fragments, device-wide atomic barrier per step)
//   K3: logits = hs·W2T + b2                   (bf16 WMMA 16x16x32, f32 acc,
//       double-buffered LDS tiles, async global->LDS for A, on-the-fly
//       fp32->bf16 of W2; operands L2-resident: 140MB < 192MB L2)
// ---------------------------------------------------------------------------

typedef float  v2f  __attribute__((ext_vector_type(2)));
typedef float  v4f  __attribute__((ext_vector_type(4)));
typedef float  v8f  __attribute__((ext_vector_type(8)));
typedef __bf16 v8bf __attribute__((ext_vector_type(8)));
typedef __bf16 v16bf __attribute__((ext_vector_type(16)));
typedef int    v4i  __attribute__((ext_vector_type(4)));

#define VOCABN 32000
#define EMBEDN 256
#define HIDN   1024
#define BN     16
#define TN     256
#define FAN1   (HIDN + EMBEDN)   // 1280
#define MTOT   (BN * TN)         // 4096

#if __has_builtin(__builtin_amdgcn_global_load_async_to_lds_b128) && \
    __has_builtin(__builtin_amdgcn_s_wait_asynccnt)
#define HAVE_ASYNC_LDS 1
#else
#define HAVE_ASYNC_LDS 0
#endif

typedef __attribute__((address_space(3))) v4i* lds_v4i_t;
typedef __attribute__((address_space(1))) v4i* glb_v4i_t;   // non-const: builtin param

static __device__ __forceinline__ void copy16_to_lds(v4i* ldst, const void* gsrc) {
#if HAVE_ASYNC_LDS
  __builtin_amdgcn_global_load_async_to_lds_b128((glb_v4i_t)gsrc, (lds_v4i_t)ldst, 0, 0);
#else
  *ldst = *(const v4i*)gsrc;
#endif
}
static __device__ __forceinline__ void wait_async_lds() {
#if HAVE_ASYNC_LDS
  __builtin_amdgcn_s_wait_asynccnt(0);
#endif
}

static __device__ __forceinline__ v8f wmma_f32_k4(v2f a, v2f b, v8f c) {
  return __builtin_amdgcn_wmma_f32_16x16x4_f32(false, a, false, b, (short)0, c, false, false);
}
static __device__ __forceinline__ v8f wmma_bf16_k32(v16bf a, v16bf b, v8f c) {
  return __builtin_amdgcn_wmma_f32_16x16x32_bf16(false, a, false, b, (short)0, c, false, false);
}

// ---------------------------------------------------------------------------
// K0: init h0 = 0 and barrier counter = 0
// ---------------------------------------------------------------------------
__global__ void k_init(float* __restrict__ hbuf, unsigned* __restrict__ ctr) {
  int i = blockIdx.x * blockDim.x + threadIdx.x;
  if (i < BN * HIDN) hbuf[i] = 0.0f;
  if (i == 0) *ctr = 0u;
}

// ---------------------------------------------------------------------------
// K1: P[(t*B+b)*H + j] = sum_e embed[input[b,t]][e] * W1[j][H+e] + b1[j]
//     One 16x16 output tile per wave; K=256 in steps of 4 (f32 WMMA).
// ---------------------------------------------------------------------------
__global__ void __launch_bounds__(256) k_embed_proj(
    const int* __restrict__ input, const float* __restrict__ embed,
    const float* __restrict__ W1, const float* __restrict__ b1,
    float* __restrict__ P) {
  const int tid  = threadIdx.x;
  const int L    = tid & 31;
  const int w    = tid >> 5;
  const int tile = blockIdx.x * 8 + w;      // 0 .. 16383
  const int mt   = tile >> 6;               // 64 N-tiles per M-tile row
  const int nt   = tile & 63;
  const int m0   = mt << 4;
  const int n0   = nt << 4;
  const int lr   = L & 15;
  const int koff = (L >> 4) << 1;           // A/B fragment K sub-offset

  const int token = input[m0 + lr];
  const float* arow = embed + (size_t)token * EMBEDN;
  const float* brow = W1 + (size_t)(n0 + lr) * FAN1 + HIDN;

  v8f acc = {0.f,0.f,0.f,0.f,0.f,0.f,0.f,0.f};
  #pragma unroll 8
  for (int kk = 0; kk < EMBEDN / 4; ++kk) {
    v2f a = *(const v2f*)(arow + (kk << 2) + koff);
    v2f b = *(const v2f*)(brow + (kk << 2) + koff);
    acc = wmma_f32_k4(a, b, acc);
  }

  const float bias  = b1[n0 + lr];
  const int   mbase = m0 + ((L >> 4) << 3);
  #pragma unroll
  for (int r = 0; r < 8; ++r) {
    int m  = mbase + r;
    int bi = m >> 8;                        // batch
    int ti = m & 255;                       // time
    P[((size_t)(ti * BN + bi) << 10) + n0 + lr] = acc[r] + bias;
  }
}

// ---------------------------------------------------------------------------
// K2: persistent scan. 32 WGs x 256 threads. WG g owns output columns
//     [32g, 32g+32). W1h slice staged once into LDS in fragment order.
//     8 waves: wave w -> tile tl=w>>2 (2 tiles), K-quarter kq=w&3.
// ---------------------------------------------------------------------------
__global__ void __launch_bounds__(256) k_scan(
    const float* __restrict__ P, const float* __restrict__ W1,
    float* __restrict__ hbuf, unsigned short* __restrict__ HSu,
    unsigned* __restrict__ ctr) {
  extern __shared__ float smem[];
  float* wfrag = smem;                       // 2*256*32*2 = 32768 floats (128KB)
  float* red   = smem + 2 * 256 * 32 * 2;    // 2*3*32*8 = 1536 floats (6KB)

  const int tid = threadIdx.x;
  const int L   = tid & 31;
  const int w   = tid >> 5;
  const int tl  = w >> 2;
  const int kq  = w & 3;
  const int n0wg = blockIdx.x << 5;
  __bf16* HS = (__bf16*)HSu;

  // Stage W1h slice into LDS, pre-swizzled to B-fragment order:
  // slot s = (tl, kk, lane): lane holds W1[n][k0 + (lane>=16?2:0) + {0,1}]
  for (int s = tid; s < 2 * 256 * 32; s += 256) {
    int stl = s >> 13;
    int skk = (s >> 5) & 255;
    int sl  = s & 31;
    int n   = n0wg + (stl << 4) + (sl & 15);
    int k   = (skk << 2) + ((sl >> 4) << 1);
    v2f v = *(const v2f*)(W1 + (size_t)n * FAN1 + k);
    *(v2f*)(wfrag + (size_t)s * 2) = v;
  }
  __syncthreads();

  const int lr    = L & 15;
  const int koff  = (L >> 4) << 1;
  const int ncol  = n0wg + (tl << 4) + lr;
  const int mhalf = (L >> 4) << 3;

  for (int t = 0; t < TN; ++t) {
    const float* hb = hbuf + ((t & 1) ? (BN * HIDN) : 0);
    float*       hn = hbuf + ((t & 1) ? 0 : (BN * HIDN));

    v8f acc = {0.f,0.f,0.f,0.f,0.f,0.f,0.f,0.f};
    const float* arow = hb + lr * HIDN;
    const int kk0 = kq << 6;
    #pragma unroll 4
    for (int kk = kk0; kk < kk0 + 64; ++kk) {
      v2f a = *(const v2f*)(arow + (kk << 2) + koff);
      v2f b = *(const v2f*)(wfrag + ((size_t)((tl << 8) + kk) * 32 + L) * 2);
      acc = wmma_f32_k4(a, b, acc);
    }

    if (kq != 0)
      *(v8f*)(red + (((tl * 3 + (kq - 1)) << 5) + L) * 8) = acc;
    __syncthreads();

    if (kq == 0) {
      #pragma unroll
      for (int q = 0; q < 3; ++q)
        acc += *(const v8f*)(red + (((tl * 3 + q) << 5) + L) * 8);
      const float* prow = P + ((size_t)(t * BN) << 10) + ncol;
      #pragma unroll
      for (int r = 0; r < 8; ++r) {
        int   m = mhalf + r;                       // batch row 0..15
        float x = acc[r] + prow[(size_t)m << 10];
        float s = 1.0f / (1.0f + __expf(-x));
        hn[m * HIDN + ncol] = s;
        HS[((size_t)(m * TN + t) << 10) + ncol] = (__bf16)s;  // hs row = b*T+t
      }
    }

    // device-wide barrier (monotonic counter, no reset race)
    __threadfence();
    __syncthreads();
    if (tid == 0) {
      __hip_atomic_fetch_add(ctr, 1u, __ATOMIC_ACQ_REL, __HIP_MEMORY_SCOPE_AGENT);
      unsigned target = (unsigned)gridDim.x * (unsigned)(t + 1);
      while (__hip_atomic_load(ctr, __ATOMIC_ACQUIRE, __HIP_MEMORY_SCOPE_AGENT) < target)
        __builtin_amdgcn_s_sleep(2);
    }
    __syncthreads();
    __threadfence();
  }
}

// ---------------------------------------------------------------------------
// K3: logits[m][n] = sum_k hs_bf16[m][k] * bf16(W2[n][k]) + b2[n]
//     WG tile 128(M) x 128(N); 8 waves = 4(M) x 2(N), each wave 32x64
//     (8 independent wmma accumulators per K-step -> XDL pipelining).
//     Double-buffered LDS; A tile staged with async global->LDS (ASYNCcnt),
//     B tile (W2) converted fp32->bf16 in regs (co-executes with WMMA).
// ---------------------------------------------------------------------------
__global__ void __launch_bounds__(256) k_logits(
    const unsigned short* __restrict__ HSu, const float* __restrict__ W2,
    const float* __restrict__ b2, float* __restrict__ out) {
  __shared__ v4i smA[2][8 * 32 * 2];   // 2 x (128x32 bf16) = 16 KB
  __shared__ v4i smB[2][8 * 32 * 2];   // 2 x (128x32 bf16) = 16 KB

  const int tid = threadIdx.x;
  const int L   = tid & 31;
  const int w   = tid >> 5;
  const int wm  = w >> 1;              // 0..3
  const int wn  = w & 1;               // 0..1
  const int m0  = blockIdx.y * 128;
  const int n0  = blockIdx.x * 128;

  v8f acc[2][4];
  #pragma unroll
  for (int i = 0; i < 2; ++i)
    #pragma unroll
    for (int j = 0; j < 4; ++j)
      acc[i][j] = (v8f){0.f,0.f,0.f,0.f,0.f,0.f,0.f,0.f};

  // stage chunk (k = kc*32) of A (bf16 copy) into buffer `buf`
  auto stageA = [&](int buf, int k) {
    #pragma unroll
    for (int c0 = 0; c0 < 2; ++c0) {
      int cc = tid + (c0 << 8);        // 512 chunks of 16B
      int mb = cc >> 6;
      int sl = (cc >> 1) & 31;
      int h  = cc & 1;
      int ko = ((sl >> 4) << 3) + (h << 4);
      const unsigned short* src =
          HSu + (size_t)(m0 + (mb << 4) + (sl & 15)) * HIDN + k + ko;
      copy16_to_lds(&smA[buf][((mb << 5) + sl) * 2 + h], src);
    }
  };
  // stage chunk of B (fp32 -> bf16 convert) into buffer `buf`
  auto stageB = [&](int buf, int k) {
    #pragma unroll
    for (int c0 = 0; c0 < 2; ++c0) {
      int cc = tid + (c0 << 8);
      int bb = cc >> 6;
      int sl = (cc >> 1) & 31;
      int h  = cc & 1;
      int ko = ((sl >> 4) << 3) + (h << 4);
      const float* src =
          W2 + (size_t)(n0 + (bb << 4) + (sl & 15)) * HIDN + k + ko;
      __builtin_prefetch((const char*)src + 256, 0, 1);  // k+2 chunk
      v4f f0 = *(const v4f*)(src);
      v4f f1 = *(const v4f*)(src + 4);
      v8bf vb;
      vb[0] = (__bf16)f0[0]; vb[1] = (__bf16)f0[1];
      vb[2] = (__bf16)f0[2]; vb[3] = (__bf16)f0[3];
      vb[4] = (__bf16)f1[0]; vb[5] = (__bf16)f1[1];
      vb[6] = (__bf16)f1[2]; vb[7] = (__bf16)f1[3];
      *(v8bf*)&smB[buf][((bb << 5) + sl) * 2 + h] = vb;
    }
  };

  stageA(0, 0);
  stageB(0, 0);
  wait_async_lds();
  __syncthreads();

  const int ab0 = wm * 2;              // A blocks ab0, ab0+1
  const int nbb = wn * 4;              // B blocks nbb .. nbb+3

  for (int kc = 0; kc < HIDN / 32; ++kc) {
    const int p = kc & 1;
    if (kc + 1 < HIDN / 32) {          // stage next chunk into other buffer
      stageA(1 - p, (kc + 1) << 5);
      stageB(1 - p, (kc + 1) << 5);
    }

    v16bf a0 = *(const v16bf*)&smA[p][(((ab0 + 0) << 5) + L) * 2];
    v16bf a1 = *(const v16bf*)&smA[p][(((ab0 + 1) << 5) + L) * 2];
    v16bf bv0 = *(const v16bf*)&smB[p][(((nbb + 0) << 5) + L) * 2];
    v16bf bv1 = *(const v16bf*)&smB[p][(((nbb + 1) << 5) + L) * 2];
    v16bf bv2 = *(const v16bf*)&smB[p][(((nbb + 2) << 5) + L) * 2];
    v16bf bv3 = *(const v16bf*)&smB[p][(((nbb + 3) << 5) + L) * 2];

    acc[0][0] = wmma_bf16_k32(a0, bv0, acc[0][0]);
    acc[0][1] = wmma_bf16_k32(a0, bv1, acc[0][1]);
    acc[0][2] = wmma_bf16_k32(a0, bv2, acc[0][2]);
    acc[0][3] = wmma_bf16_k32(a0, bv3, acc[0][3]);
    acc[1][0] = wmma_bf16_k32(a1, bv0, acc[1][0]);
    acc[1][1] = wmma_bf16_k32(a1, bv1, acc[1][1]);
    acc[1][2] = wmma_bf16_k32(a1, bv2, acc[1][2]);
    acc[1][3] = wmma_bf16_k32(a1, bv3, acc[1][3]);

    wait_async_lds();
    __syncthreads();
  }

  const int lr = L & 15;
  const int mh = (L >> 4) << 3;
  #pragma unroll
  for (int i = 0; i < 2; ++i) {
    #pragma unroll
    for (int j = 0; j < 4; ++j) {
      int   nn   = n0 + ((nbb + j) << 4) + lr;
      float bias = b2[nn];
      int   mb   = m0 + ((ab0 + i) << 4) + mh;
      #pragma unroll
      for (int r = 0; r < 8; ++r)
        out[(size_t)(mb + r) * VOCABN + nn] = acc[i][j][r] + bias;
    }
  }
}

// ---------------------------------------------------------------------------
// launch
// ---------------------------------------------------------------------------
extern "C" void kernel_launch(void* const* d_in, const int* in_sizes, int n_in,
                              void* d_out, int out_size, void* d_ws, size_t ws_size,
                              hipStream_t stream) {
  const int*   input = (const int*)d_in[0];     // [16,256] int32
  const float* embed = (const float*)d_in[1];   // [32000,256]
  const float* W1    = (const float*)d_in[2];   // [1024,1280]
  const float* b1    = (const float*)d_in[3];   // [1024]
  const float* W2    = (const float*)d_in[4];   // [32000,1024]
  const float* b2    = (const float*)d_in[5];   // [32000]
  float*       out   = (float*)d_out;           // [16,256,32000]

  // workspace layout (~25.3 MB)
  char* ws = (char*)d_ws;
  unsigned*       ctr  = (unsigned*)ws;
  float*          hbuf = (float*)(ws + 1024);                       // 2*16*1024 f32
  float*          P    = (float*)(ws + 1024 + 2 * BN * HIDN * 4);   // 4096*1024 f32
  unsigned short* HS   = (unsigned short*)(ws + 1024 + 2 * BN * HIDN * 4
                                           + (size_t)MTOT * HIDN * 4); // bf16 hs

  k_init<<<64, 256, 0, stream>>>(hbuf, ctr);
  k_embed_proj<<<2048, 256, 0, stream>>>(input, embed, W1, b1, P);
  // dynamic LDS: 128KB W1h fragments + 6KB reduction
  k_scan<<<32, 256, (2 * 256 * 32 * 2 + 2 * 3 * 32 * 8) * sizeof(float), stream>>>(
      P, W1, hbuf, HS, ctr);
  k_logits<<<dim3(VOCABN / 128, MTOT / 128), 256, 0, stream>>>(HS, W2, b2, out);
}
